// LSTMInnerRecLayer_18305150616457
// MI455X (gfx1250) — compile-verified, matrix operand order checked by reading, not run on hardware
//
#include <hip/hip_runtime.h>
#include <hip/hip_bf16.h>
#include <math.h>

// ---------------------------------------------------------------------------
// LSTM inner recurrence for MI455X (gfx1250, wave32, WMMA + async-to-LDS).
//   gates = h @ W_hh^T + (b_ih + b_hh)   (input path is identically zero)
//   c = sigmoid(f)*c + sigmoid(i)*tanh(g);  h = sigmoid(o)*tanh(c)
// T=512 sequential fused step-kernels. W_hh held as bf16 (8MB, L2-resident).
// Per step: double-buffered GLOBAL_LOAD_ASYNC_TO_LDS staging of A(h) and
// B(W_hh) K-chunks, bank-conflict-free ds_load fragments, fp32 accumulation
// via v_wmma_f32_16x16x32_bf16.
// ---------------------------------------------------------------------------

typedef __attribute__((ext_vector_type(16))) __bf16 v16bf;
typedef __attribute__((ext_vector_type(8)))  __bf16 v8bf;
typedef __attribute__((ext_vector_type(8)))  float  v8f;

#define LSTM_T 512
#define LSTM_B 128
#define LSTM_H 1024
#define LSTM_G 4096       // 4*H
#define KC     64         // K-chunk per LDS stage
#define KROW   72         // padded LDS row stride (elements): 36-bank row skew
#define NCHUNK (LSTM_H / KC)

// ---- CDNA5 async global->LDS copy (16B per lane, tracked by ASYNCcnt) -----
__device__ __forceinline__ void async_copy_b128(uint32_t lds_addr, const void* gaddr) {
    // VDST = LDS address VGPR, VADDR = 64-bit global address (GV mode).
    asm volatile("global_load_async_to_lds_b128 %0, %1, off"
                 :: "v"(lds_addr), "v"(gaddr) : "memory");
}
__device__ __forceinline__ void wait_async_8() {
    asm volatile("s_wait_asynccnt 0x8" ::: "memory");
}
__device__ __forceinline__ void wait_async_0() {
    asm volatile("s_wait_asynccnt 0x0" ::: "memory");
}

// Load a 16-element bf16 WMMA fragment slice for this lane (from LDS).
// CDNA5 16-bit A/B layout: lanes 0-15 hold K={k0..k0+7, k0+16..k0+23},
// lanes 16-31 hold K={k0+8..k0+15, k0+24..k0+31}; lane-half +8 baked by caller.
__device__ __forceinline__ v16bf load_frag16(const __bf16* base) {
    union { v16bf v; v8bf h[2]; } u;
    u.h[0] = *reinterpret_cast<const v8bf*>(base);
    u.h[1] = *reinterpret_cast<const v8bf*>(base + 16);
    return u.v;
}

__device__ __forceinline__ float fast_sigmoid(float x) {
    return 1.0f / (1.0f + __expf(-x));
}

// ---- prep kernels ---------------------------------------------------------

__global__ __launch_bounds__(256) void prep_weights(const float* __restrict__ W,
                                                    __bf16* __restrict__ Wb, int n) {
    int i = blockIdx.x * blockDim.x + threadIdx.x;
    if (i < n) Wb[i] = (__bf16)W[i];
}

__global__ __launch_bounds__(256) void prep_state(const float* __restrict__ h0,
                                                  const float* __restrict__ c0,
                                                  const float* __restrict__ b_ih,
                                                  const float* __restrict__ b_hh,
                                                  __bf16* __restrict__ h_bf,
                                                  float* __restrict__ c_buf,
                                                  float* __restrict__ bsum, int n) {
    int i = blockIdx.x * blockDim.x + threadIdx.x;
    if (i < n) {
        h_bf[i]  = (__bf16)h0[i];
        c_buf[i] = c0[i];
    }
    if (i < LSTM_G) bsum[i] = b_ih[i] + b_hh[i];
}

// ---- fused LSTM step ------------------------------------------------------
// Grid: 32 blocks; block owns 32 h-columns x all 128 rows x all 4 gates.
// Block: 256 threads = 8 wave32; wave w owns rows [16w, 16w+16).
// Per wave: 4 gates x 2 col-tiles = 8 accumulator tiles, K=1024.

__global__ __launch_bounds__(256) void lstm_step(const __bf16* __restrict__ Wb,    // [4096,1024] bf16
                                                 const float*  __restrict__ bsum,  // [4096]
                                                 const __bf16* __restrict__ h_in,  // [128,1024] bf16
                                                 __bf16*       __restrict__ h_out, // [128,1024] bf16
                                                 float*        __restrict__ c_buf, // [128,1024]
                                                 float*        __restrict__ outs_t)// d_out + t*B*H
{
    // Double-buffered LDS stages: A = h rows [128 x KC], B = W tile [128 x KC],
    // rows padded 64->72 elements (36-bank skew -> conflict-free b128 reads).
    __shared__ __align__(16) __bf16 sA[2][128 * KROW];
    __shared__ __align__(16) __bf16 sB[2][128 * KROW];

    const int tid    = threadIdx.x;
    const int lane   = tid & 31;
    const int wave   = tid >> 5;             // 0..7 -> row tile
    const int lanelo = lane & 15;
    const int lanehi = lane >> 4;            // K-half select
    const int rowbase = wave * 16;
    const int colbase = blockIdx.x * 32;     // 32 h-columns per block

    // Accumulators, bias pre-folded (C tile: col depends only on lanelo).
    v8f acc[4][2];
#pragma unroll
    for (int g = 0; g < 4; ++g)
#pragma unroll
        for (int ct = 0; ct < 2; ++ct) {
            const float bv = bsum[g * LSTM_H + colbase + ct * 16 + lanelo];
#pragma unroll
            for (int j = 0; j < 8; ++j) acc[g][ct][j] = bv;
        }

    // Cooperative async stage of one K-chunk into buffer nb.
    // 16KB per matrix per chunk = 1024 b128 vectors; 4 per thread per matrix.
    auto issue_chunk = [&](int ch, int nb) {
        const int kc = ch * KC;
#pragma unroll
        for (int j = 0; j < 4; ++j) {
            const int v   = tid + j * 256;   // 0..1023
            const int row = v >> 3;          // tile row 0..127 (8 vecs/row)
            const int kv  = v & 7;           // 16B vector within row
            // A: h rows 0..127 directly.
            const __bf16* srcA = h_in + (size_t)row * LSTM_H + kc + kv * 8;
            const uint32_t dstA = (uint32_t)(uintptr_t)&sA[nb][row * KROW + kv * 8];
            async_copy_b128(dstA, srcA);
            // B: tile row -> W row (gate g, local col lc); contiguous in k.
            const int g  = row >> 5;
            const int lc = row & 31;
            const __bf16* srcB = Wb + (size_t)(g * LSTM_H + colbase + lc) * LSTM_H + kc + kv * 8;
            const uint32_t dstB = (uint32_t)(uintptr_t)&sB[nb][row * KROW + kv * 8];
            async_copy_b128(dstB, srcB);
        }
    };

    issue_chunk(0, 0);

    for (int ch = 0; ch < NCHUNK; ++ch) {
        const int cb = ch & 1;
        if (ch + 1 < NCHUNK) {
            issue_chunk(ch + 1, (ch + 1) & 1);  // prefetch next chunk (other buffer)
            wait_async_8();                      // chunk ch's 8 ops complete
        } else {
            wait_async_0();
        }
        __syncthreads();                         // chunk ch visible to all waves

        // 2 K-steps of 32; A fragment reused by 8 WMMA tiles each.
#pragma unroll
        for (int ks = 0; ks < 2; ++ks) {
            const int kk = ks * 32 + lanehi * 8;
            const v16bf a = load_frag16(&sA[cb][(rowbase + lanelo) * KROW + kk]);
#pragma unroll
            for (int g = 0; g < 4; ++g)
#pragma unroll
                for (int ct = 0; ct < 2; ++ct) {
                    const v16bf b =
                        load_frag16(&sB[cb][(g * 32 + ct * 16 + lanelo) * KROW + kk]);
                    acc[g][ct] = __builtin_amdgcn_wmma_f32_16x16x32_bf16(
                        false, a, false, b, (short)0, acc[g][ct], false, false);
                }
        }
        __syncthreads();                         // done reading cb before it is refilled
    }

    // Epilogue: gate nonlinearities + state update, fully lane-local.
    // C/D tile layout: VGPR v, lanes 0-15 -> M=v, lanes 16-31 -> M=v+8; N=lanelo.
#pragma unroll
    for (int ct = 0; ct < 2; ++ct) {
        const int hcol = colbase + ct * 16 + lanelo;
#pragma unroll
        for (int v = 0; v < 8; ++v) {
            const int row = rowbase + v + lanehi * 8;
            const float ig = fast_sigmoid(acc[0][ct][v]);
            const float fg = fast_sigmoid(acc[1][ct][v]);
            const float gg = tanhf(acc[2][ct][v]);
            const float og = fast_sigmoid(acc[3][ct][v]);
            const size_t idx = (size_t)row * LSTM_H + hcol;
            const float c = fg * c_buf[idx] + ig * gg;
            c_buf[idx] = c;
            const float h = og * tanhf(c);
            outs_t[idx] = h;
            h_out[idx] = (__bf16)h;
        }
    }
}

// ---- launch ---------------------------------------------------------------

extern "C" void kernel_launch(void* const* d_in, const int* in_sizes, int n_in,
                              void* d_out, int out_size, void* d_ws, size_t ws_size,
                              hipStream_t stream) {
    // setup_inputs order: input, h0, c0, W_ih, W_hh, b_ih, b_hh
    const float* h0   = (const float*)d_in[1];
    const float* c0   = (const float*)d_in[2];
    const float* W_hh = (const float*)d_in[4];
    const float* b_ih = (const float*)d_in[5];
    const float* b_hh = (const float*)d_in[6];
    float* outs = (float*)d_out;                          // [T,B,H] then h[B,H], c[B,H]

    // Workspace layout (256B-aligned slices).
    char* ws = (char*)d_ws;
    size_t off = 0;
    auto take = [&](size_t bytes) { char* p = ws + off; off = (off + bytes + 255) & ~(size_t)255; return p; };
    __bf16* Wb    = (__bf16*)take((size_t)LSTM_G * LSTM_H * sizeof(__bf16)); // 8 MB
    __bf16* h_bf0 = (__bf16*)take((size_t)LSTM_B * LSTM_H * sizeof(__bf16)); // 256 KB
    __bf16* h_bf1 = (__bf16*)take((size_t)LSTM_B * LSTM_H * sizeof(__bf16)); // 256 KB
    float*  c_buf = (float*) take((size_t)LSTM_B * LSTM_H * sizeof(float));  // 512 KB
    float*  bsum  = (float*) take((size_t)LSTM_G * sizeof(float));           // 16 KB

    // 1) per-call prep: bf16 weights, bias sum, initial state.
    {
        const int nW = LSTM_G * LSTM_H;
        prep_weights<<<(nW + 255) / 256, 256, 0, stream>>>(W_hh, Wb, nW);
        const int nS = LSTM_B * LSTM_H;
        prep_state<<<(nS + 255) / 256, 256, 0, stream>>>(h0, c0, b_ih, b_hh,
                                                         h_bf0, c_buf, bsum, nS);
    }

    // 2) T sequential fused steps (stream order provides the dependency chain).
    __bf16* hb[2] = { h_bf0, h_bf1 };
    for (int t = 0; t < LSTM_T; ++t) {
        lstm_step<<<LSTM_H / 32, 256, 0, stream>>>(
            Wb, bsum, hb[t & 1], hb[(t + 1) & 1], c_buf,
            outs + (size_t)t * LSTM_B * LSTM_H);
    }

    // 3) final h = outs[T-1]; final c = c_buf (graph-safe D2D copies).
    float* h_final = outs + (size_t)LSTM_T * LSTM_B * LSTM_H;
    float* c_final = h_final + (size_t)LSTM_B * LSTM_H;
    hipMemcpyAsync(h_final, outs + (size_t)(LSTM_T - 1) * LSTM_B * LSTM_H,
                   (size_t)LSTM_B * LSTM_H * sizeof(float), hipMemcpyDeviceToDevice, stream);
    hipMemcpyAsync(c_final, c_buf,
                   (size_t)LSTM_B * LSTM_H * sizeof(float), hipMemcpyDeviceToDevice, stream);
}
